// EuclideanTransformer_61546881352454
// MI455X (gfx1250) — matrix operand order repaired and checked by reading, not compiled, not run on hardware
//
#include <hip/hip_runtime.h>
#include <hip/hip_bf16.h>
#include <stdint.h>

// ---------------------------------------------------------------------------
// Problem constants (match reference). NN and EE are multiples of 16, so
// 16-row strips tile both exactly (no row guards needed anywhere).
// ---------------------------------------------------------------------------
#define NN 50000
#define EE 400000
#define RR 32
#define FF 132
#define EVD 15

typedef __attribute__((ext_vector_type(16))) _Float16 v16h;
typedef __attribute__((ext_vector_type(8)))  _Float16 v8h;
typedef __attribute__((ext_vector_type(4)))  _Float16 v4h;
typedef __attribute__((ext_vector_type(8)))  float    v8f;
typedef __attribute__((ext_vector_type(4)))  unsigned int u32x4;
typedef __attribute__((ext_vector_type(8)))  int          i32x8;
typedef __attribute__((ext_vector_type(4)))  int          i32x4;

#if defined(__has_builtin)
#if __has_builtin(__builtin_amdgcn_tensor_load_to_lds) && \
    __has_builtin(__builtin_amdgcn_s_wait_tensorcnt)
#define HAVE_TDM 1
#endif
#endif
#ifndef HAVE_TDM
#define HAVE_TDM 0
#endif

__device__ inline float fast_rcp(float x) {
#if defined(__has_builtin) && __has_builtin(__builtin_amdgcn_rcpf)
    return __builtin_amdgcn_rcpf(x);
#else
    return 1.0f / x;
#endif
}
__device__ inline float silu_f(float x) {
    return x * fast_rcp(1.0f + __expf(-x));  // v_exp + v_rcp, no IEEE divide
}

__device__ inline v8f wmma16(v16h a, v16h b, v8f c) {
    // D = A(16x32 f16) x B(32x16 f16) + C(16x16 f32)
    return __builtin_amdgcn_wmma_f32_16x16x32_f16(false, a, false, b, (short)0, c,
                                                  false, false);
}

// ---------------------------------------------------------------------------
// Fragment helpers (wave32, ISA 7.12.2 layouts)
// ---------------------------------------------------------------------------
// A fragment from row-major storage (ld halves, 16B-aligned rows): per lane two
// contiguous 8-half runs -> two b128 loads.
__device__ inline v16h load_a16(const _Float16* tile, int ld) {
    const int lane = threadIdx.x & 31;
    const int row  = lane & 15;
    const int kb   = (lane >> 4) << 3;
    const _Float16* p = tile + row * ld + kb;
    v8h lo = *(const v8h*)(p);
    v8h hi = *(const v8h*)(p + 16);
    return __builtin_shufflevector(lo, hi, 0, 1, 2, 3, 4, 5, 6, 7, 8, 9, 10, 11, 12,
                                   13, 14, 15);
}

// B fragment from pre-swizzled storage: lane's 16 halves contiguous.
__device__ inline v16h load_b_swz(const _Float16* tile) {
    const _Float16* p = tile + (threadIdx.x & 31) * 16;
    v8h lo = *(const v8h*)(p);
    v8h hi = *(const v8h*)(p + 8);
    return __builtin_shufflevector(lo, hi, 0, 1, 2, 3, 4, 5, 6, 7, 8, 9, 10, 11, 12,
                                   13, 14, 15);
}

// D store with per-lane-uniform column guard hoisted out of the element loop.
// f(row, col, value); executes only when col < colMax.
template <typename F>
__device__ inline void store_d_guard_col(const v8f& d, int colBase, int colMax,
                                         F&& f) {
    const int lane = threadIdx.x & 31;
    const int col  = colBase + (lane & 15);
    const int rb   = (lane >> 4) << 3;
    if (col < colMax) {
#pragma unroll
        for (int i = 0; i < 8; ++i) f(rb + i, col, d[i]);
    }
}

// ---------------------------------------------------------------------------
// Workspace layout constants (f16 halves). B matrices stored swizzled:
// [kt][nt][lane][16], tile = 512 halves.
// ---------------------------------------------------------------------------
#define EW_RW1_OFF 0
#define EW_RW2_OFF (32 * 144)                 // 4608   (KT=1, NT=9)
#define EW_EW1_OFF (EW_RW2_OFF + 160 * 144)   // 27648  (KT=5, NT=9)
#define EW_EW2_OFF (EW_EW1_OFF + 32 * 48)     // 29184  (KT=1, NT=3)
#define EW_BLK     (EW_EW2_OFF + 64 * 144)    // 38400 halves per filter net

// ---------------------------------------------------------------------------
// Packing kernels
// ---------------------------------------------------------------------------
__global__ void k_pack_b_swz(const float* __restrict__ src, _Float16* __restrict__ dst,
                             int Ksrc, int Nsrc, int KT, int NT, int batch,
                             int srcBatchStride) {
    int idx = blockIdx.x * blockDim.x + threadIdx.x;
    int tot = batch * KT * NT * 512;
    if (idx >= tot) return;
    int b    = idx / (KT * NT * 512);
    int r    = idx - b * (KT * NT * 512);
    int tile = r >> 9;
    int e    = r & 511;
    int lane = e >> 4, j = e & 15;
    int kt = tile / NT, nt = tile - kt * NT;
    int k = kt * 32 + ((j & 8) << 1) + ((lane >> 4) << 3) + (j & 7);
    int n = nt * 16 + (lane & 15);
    float v = (k < Ksrc && n < Nsrc) ? src[b * srcBatchStride + k * Nsrc + n] : 0.0f;
    dst[idx] = (_Float16)v;
}

// Padded per-head f16 node features: dst[n][h*64+k] = (k<D) ? xin[n*FF+h*D+k] : 0
__global__ void k_pack_x(const float* __restrict__ xin, _Float16* __restrict__ dst,
                         int H, int D) {
    int idx = blockIdx.x * blockDim.x + threadIdx.x;
    int tot = NN * H * 64;
    if (idx >= tot) return;
    int n = idx / (H * 64), r = idx - n * (H * 64);
    int h = r >> 6, k = r & 63;
    dst[idx] = (_Float16)((k < D) ? xin[n * FF + h * D + k] : 0.0f);
}

__global__ void k_zero(float* __restrict__ p, long long n) {
    long long i = (long long)blockIdx.x * blockDim.x + threadIdx.x;
    long long s = (long long)gridDim.x * blockDim.x;
    for (; i < n; i += s) p[i] = 0.0f;
}

// ---------------------------------------------------------------------------
// Node projections: q/k/v_inv (4 heads x 33) and q/k_ev (3 heads x 44)
// ---------------------------------------------------------------------------
__global__ void __launch_bounds__(256) k_node_proj(
    const _Float16* __restrict__ xi_inv,   // [NN][4*64] padded f16
    const _Float16* __restrict__ xi_ev,    // [NN][3*64] padded f16
    const _Float16* __restrict__ wqkv_inv, const _Float16* __restrict__ wqk_ev,
    float* __restrict__ q_inv, float* __restrict__ k_inv, float* __restrict__ v_inv,
    float* __restrict__ q_ev, float* __restrict__ k_ev) {
    const int wave  = threadIdx.x >> 5;
    const int strip = blockIdx.x * 8 + wave;
    const int n0    = strip * 16;
    if (n0 >= NN) return;  // wave-uniform

    // ----- invariant heads: x(16x33 pad64) @ W(64x48) -----
    for (int set = 0; set < 3; ++set) {
        float* out = (set == 0) ? q_inv : (set == 1) ? k_inv : v_inv;
        for (int h = 0; h < 4; ++h) {
            const _Float16* W  = wqkv_inv + (set * 4 + h) * (64 * 48);  // KT=2,NT=3
            const _Float16* xh = xi_inv + (size_t)n0 * 256 + h * 64;
            v16h a0 = load_a16(xh, 256);
            v16h a1 = load_a16(xh + 32, 256);
#pragma unroll
            for (int t = 0; t < 3; ++t) {
                v8f acc = {};
                acc = wmma16(a0, load_b_swz(W + (0 * 3 + t) * 512), acc);
                acc = wmma16(a1, load_b_swz(W + (1 * 3 + t) * 512), acc);
                store_d_guard_col(acc, t * 16, 33, [&](int r, int col, float v) {
                    out[(n0 + r) * FF + h * 33 + col] = (set == 2) ? v : silu_f(v);
                });
            }
        }
    }
    // ----- equivariant heads: x(16x44 pad64) @ W(64x48) -----
    for (int set = 0; set < 2; ++set) {
        float* out = (set == 0) ? q_ev : k_ev;
        for (int h = 0; h < 3; ++h) {
            const _Float16* W  = wqk_ev + (set * 3 + h) * (64 * 48);
            const _Float16* xh = xi_ev + (size_t)n0 * 192 + h * 64;
            v16h a0 = load_a16(xh, 192);
            v16h a1 = load_a16(xh + 32, 192);
#pragma unroll
            for (int t = 0; t < 3; ++t) {
                v8f acc = {};
                acc = wmma16(a0, load_b_swz(W + (0 * 3 + t) * 512), acc);
                acc = wmma16(a1, load_b_swz(W + (1 * 3 + t) * 512), acc);
                store_d_guard_col(acc, t * 16, 44, [&](int r, int col, float v) {
                    out[(n0 + r) * FF + h * 44 + col] = silu_f(v);
                });
            }
        }
    }
}

// ---------------------------------------------------------------------------
// Edge kernels: filter net (WMMA on LDS-staged weights) + attention + scatter
// ---------------------------------------------------------------------------
struct alignas(16) EdgeLds {
    _Float16 wrW1[32 * 144];   // swizzled B tiles (KT=1, NT=9)
    _Float16 wrW2[160 * 144];  // swizzled B tiles (KT=5, NT=9)
    _Float16 weW1[32 * 48];    // swizzled B tiles (KT=1, NT=3)
    _Float16 weW2[64 * 144];   // swizzled B tiles (KT=2, NT=9)
    _Float16 rbfA[16 * 32];    // A tile: 16 edges x 32 rbf (row-major)
    _Float16 evA[16 * 32];     // A tile: 16 edges x (3 real + zero pad)
    _Float16 h1[16 * 160];     // hidden activations (pad cols stay zero)
    _Float16 he[16 * 64];      // ev hidden (pad cols stay zero)
    float    fw[16 * 144];     // filter output (f32); pad cols unused
    float    cutoff[16];
    float    alpha[16 * 4];
    int      sl[16];
    int      rl[16];
};

__device__ inline void edge_prologue(EdgeLds& L, const _Float16* __restrict__ wblk) {
    const int tid = threadIdx.x;
#if HAVE_TDM
    // Stage all filter-net weights with the Tensor Data Mover: one 1-D tile of
    // 19200 4-byte elements (= EW_BLK halves), global -> LDS offset 0.
    if (tid < 32) {
        uint64_t ga = (uint64_t)(uintptr_t)wblk;
        u32x4 g0;
        g0[0] = 1u;                                  // count=1, user descriptor
        g0[1] = 0u;                                  // lds_addr = 0
        g0[2] = (unsigned)(ga & 0xffffffffu);        // global_addr[31:0]
        g0[3] = (unsigned)((ga >> 32) & 0x01ffffffu) // global_addr[56:32]
                | 0x80000000u;                       // type = 2 ("image")
        i32x8 g1;
        g1[0] = (int)(2u << 16);                     // data_size = 4 bytes
        g1[1] = (int)(19200u << 16);                 // tensor_dim0[15:0]
        g1[2] = (int)(1u << 16);                     // dim0 hi=0 | tensor_dim1=1
        g1[3] = (int)(19200u << 16);                 // dim1 hi=0 | tile_dim0
        g1[4] = 1;                                   // tile_dim1=1, tile_dim2=0
        g1[5] = 19200;                               // tensor_dim0_stride lo
        g1[6] = 0;
        g1[7] = 0;
        i32x4 gz = {0, 0, 0, 0};
#if __clang_major__ >= 23
        i32x8 gz8 = {0, 0, 0, 0, 0, 0, 0, 0};
        __builtin_amdgcn_tensor_load_to_lds(g0, g1, gz, gz, gz8, 0);
#else
        __builtin_amdgcn_tensor_load_to_lds(g0, g1, gz, gz, 0);
#endif
        __builtin_amdgcn_s_wait_tensorcnt(0);
    }
#else
    {
        const uint4* s = (const uint4*)wblk;
        uint4*       d = (uint4*)&L;  // struct begins with the weight arrays
        for (int i = tid; i < (int)(EW_BLK * 2 / 16); i += 128) d[i] = s[i];
    }
#endif
    // One-time zero of the activation region (evA..he contiguous: 512+2560+1024
    // halves = 2048 dwords). Pad columns are never re-written afterwards, so
    // every WMMA sees finite (zero) pad values.
    {
        uint32_t* z = (uint32_t*)L.evA;
        for (int i = tid; i < (512 + 2560 + 1024) / 2; i += 128) z[i] = 0u;
    }
    __syncthreads();
}

__device__ inline void edge_common_setup(EdgeLds& L, int e0,
                                         const float* __restrict__ rbf,
                                         const float* __restrict__ cutoffs,
                                         const int* __restrict__ senders,
                                         const int* __restrict__ receivers,
                                         const float* __restrict__ evf) {
    const int tid = threadIdx.x;
    // rbf tile: float4 loads, packed 4-half stores (16 edges x 32)
    {
        const float4* src = (const float4*)(rbf + (size_t)e0 * RR);
        int t = tid >> 3, q = tid & 7;  // t = edge row, q = quad of 4
        float4 v = src[t * 8 + q];
        v4h  h;
        h[0] = (_Float16)v.x; h[1] = (_Float16)v.y;
        h[2] = (_Float16)v.z; h[3] = (_Float16)v.w;
        *(v4h*)(&L.rbfA[t * 32 + q * 4]) = h;
    }
    if (tid < 16) {
        int e = e0 + tid;
        int s = senders[e];
        int r = receivers[e];
        L.sl[tid]     = s;
        L.rl[tid]     = r;
        L.cutoff[tid] = cutoffs[e];
        float g0 = 0.f, g1 = 0.f, g2 = 0.f;
        for (int j = 0; j < EVD; ++j) {
            float d  = evf[s * EVD + j] - evf[r * EVD + j];
            float dd = d * d;
            if (j < 3) g0 += dd;
            else if (j < 8) g1 += dd;
            else g2 += dd;
        }
        L.evA[tid * 32 + 0] = (_Float16)g0;
        L.evA[tid * 32 + 1] = (_Float16)g1;
        L.evA[tid * 32 + 2] = (_Float16)g2;
    }
}

// fw = (silu(rbf@rW1+rb1))@rW2 + (silu(ev3@eW1+eb1))@eW2 + rb2 + eb2
__device__ inline void filter_net_16(EdgeLds& L, const float* __restrict__ rb1,
                                     const float* __restrict__ rb2,
                                     const float* __restrict__ eb1,
                                     const float* __restrict__ eb2) {
    const int wave = threadIdx.x >> 5;
    // layer 1: rbf path (K = 32, single k-step); A fragment hoisted per wave
    {
        v16h ar = load_a16(L.rbfA, 32);
        for (int t = wave; t < 9; t += 4) {
            v8f acc = {};
            acc     = wmma16(ar, load_b_swz(L.wrW1 + t * 512), acc);
            store_d_guard_col(acc, t * 16, FF, [&](int r, int col, float v) {
                L.h1[r * 160 + col] = (_Float16)silu_f(v + rb1[col]);
            });
        }
        v16h ae = load_a16(L.evA, 32);
        for (int t = wave; t < 3; t += 4) {
            v8f acc = {};
            acc     = wmma16(ae, load_b_swz(L.weW1 + t * 512), acc);
            store_d_guard_col(acc, t * 16, 33, [&](int r, int col, float v) {
                L.he[r * 64 + col] = (_Float16)silu_f(v + eb1[col]);
            });
        }
    }
    __syncthreads();
    // layer 2: A fragments loaded once per wave, reused across output tiles
    v16h ah[5], ae2[2];
#pragma unroll
    for (int ks = 0; ks < 5; ++ks) ah[ks] = load_a16(L.h1 + ks * 32, 160);
#pragma unroll
    for (int ks = 0; ks < 2; ++ks) ae2[ks] = load_a16(L.he + ks * 32, 64);
    for (int t = wave; t < 9; t += 4) {
        v8f acc = {};
#pragma unroll
        for (int ks = 0; ks < 5; ++ks)
            acc = wmma16(ah[ks], load_b_swz(L.wrW2 + (ks * 9 + t) * 512), acc);
#pragma unroll
        for (int ks = 0; ks < 2; ++ks)
            acc = wmma16(ae2[ks], load_b_swz(L.weW2 + (ks * 9 + t) * 512), acc);
        store_d_guard_col(acc, t * 16, FF, [&](int r, int col, float v) {
            L.fw[r * 144 + col] = v + rb2[col] + eb2[col];
        });
    }
    __syncthreads();
}

__global__ void __launch_bounds__(128) k_edge_inv(
    const float* __restrict__ rbf, const float* __restrict__ cutoffs,
    const int* __restrict__ senders, const int* __restrict__ receivers,
    const float* __restrict__ evf, const _Float16* __restrict__ wblk,
    const float* __restrict__ rb1, const float* __restrict__ rb2,
    const float* __restrict__ eb1, const float* __restrict__ eb2,
    const float* __restrict__ q_inv, const float* __restrict__ k_inv,
    const float* __restrict__ v_inv, float* __restrict__ d_inv) {
    extern __shared__ char smem[];
    EdgeLds& L    = *reinterpret_cast<EdgeLds*>(smem);
    const int tid = threadIdx.x;
    edge_prologue(L, wblk);

    const int nStrips = EE / 16;
    for (int strip = blockIdx.x; strip < nStrips; strip += gridDim.x) {
        const int e0 = strip * 16;
        {   // prefetch next strip's rbf rows into cache
            int en = e0 + gridDim.x * 16;
            if (en < EE) __builtin_prefetch(rbf + (size_t)en * RR, 0, 1);
        }
        edge_common_setup(L, e0, rbf, cutoffs, senders, receivers, evf);
        __syncthreads();
        filter_net_16(L, rb1, rb2, eb1, eb2);
        // alpha_inv[e][h] = cutoff * sum_d q[r]*k[s]*fw / sqrt(33)
        if (tid < 64) {
            int e = tid >> 2, h = tid & 3;
            int s = L.sl[e], r = L.rl[e];
            float sum = 0.0f;
            for (int d = 0; d < 33; ++d)
                sum += q_inv[r * FF + h * 33 + d] * k_inv[s * FF + h * 33 + d] *
                       L.fw[e * 144 + h * 33 + d];
            L.alpha[e * 4 + h] = sum * L.cutoff[e] * 0.17407766f;  // 1/sqrt(33)
        }
        __syncthreads();
        // scatter msg_inv: 8 threads per edge
        {
            int e = tid >> 3, s = 0, r = 0;
            e &= 15;
            s = L.sl[e]; r = L.rl[e];
            for (int f = (tid & 7); f < FF; f += 8) {
                int   h = (f >= 99) ? 3 : (f >= 66) ? 2 : (f >= 33) ? 1 : 0;
                float v = L.alpha[e * 4 + h] * v_inv[s * FF + f];
                atomicAdd(&d_inv[r * FF + f], v);
            }
        }
        __syncthreads();
    }
}

__global__ void __launch_bounds__(128) k_edge_ev(
    const float* __restrict__ rbf, const float* __restrict__ cutoffs,
    const int* __restrict__ senders, const int* __restrict__ receivers,
    const float* __restrict__ evf, const float* __restrict__ sh,
    const _Float16* __restrict__ wblk, const float* __restrict__ rb1,
    const float* __restrict__ rb2, const float* __restrict__ eb1,
    const float* __restrict__ eb2, const float* __restrict__ q_ev,
    const float* __restrict__ k_ev, float* __restrict__ d_ev) {
    extern __shared__ char smem[];
    EdgeLds& L    = *reinterpret_cast<EdgeLds*>(smem);
    const int tid = threadIdx.x;
    edge_prologue(L, wblk);

    const int nStrips = EE / 16;
    for (int strip = blockIdx.x; strip < nStrips; strip += gridDim.x) {
        const int e0 = strip * 16;
        {
            int en = e0 + gridDim.x * 16;
            if (en < EE) __builtin_prefetch(rbf + (size_t)en * RR, 0, 1);
        }
        edge_common_setup(L, e0, rbf, cutoffs, senders, receivers, evf);
        __syncthreads();
        filter_net_16(L, rb1, rb2, eb1, eb2);
        // alpha_ev[e][h] (3 heads x 44)
        if (tid < 48) {
            int e = tid / 3, h = tid - e * 3;
            int s = L.sl[e], r = L.rl[e];
            float sum = 0.0f;
            for (int d = 0; d < 44; ++d)
                sum += q_ev[r * FF + h * 44 + d] * k_ev[s * FF + h * 44 + d] *
                       L.fw[e * 144 + h * 44 + d];
            L.alpha[e * 4 + h] = sum * L.cutoff[e] * 0.15075567f;  // 1/sqrt(44)
        }
        __syncthreads();
        // scatter msg_ev = repeat(alpha)*sh: 8 threads per edge
        {
            int e = (tid >> 3) & 15;
            int r = L.rl[e];
            for (int j = (tid & 7); j < EVD; j += 8) {
                int   g = (j < 3) ? 0 : (j < 8) ? 1 : 2;
                float v = L.alpha[e * 4 + g] * sh[(size_t)(e0 + e) * EVD + j];
                atomicAdd(&d_ev[r * EVD + j], v);
            }
        }
        __syncthreads();
    }
}

// ---------------------------------------------------------------------------
// Finish 1a: coalesced inv residual + f16 concat staging
// ---------------------------------------------------------------------------
__global__ void k_finish1a(const float* __restrict__ xin,
                           const float* __restrict__ d_inv, float* __restrict__ inv1,
                           _Float16* __restrict__ cat_h) {
    int i = blockIdx.x * blockDim.x + threadIdx.x;
    if (i >= NN * FF) return;
    int n = i / FF, f = i - n * FF;
    float v = xin[i] + d_inv[i];
    inv1[i] = v;
    cat_h[(size_t)n * 160 + f] = (_Float16)v;
}

// Finish 1b: per-node ev residual, ev1^2 group sums, concat tail
__global__ void k_finish1b(const float* __restrict__ evin,
                           const float* __restrict__ d_ev, float* __restrict__ ev1,
                           _Float16* __restrict__ cat_h) {
    int n = blockIdx.x * blockDim.x + threadIdx.x;
    if (n >= NN) return;
    float g0 = 0.f, g1 = 0.f, g2 = 0.f;
    for (int j = 0; j < EVD; ++j) {
        float v          = evin[n * EVD + j] + d_ev[n * EVD + j];
        ev1[n * EVD + j] = v;
        if (j < 3) g0 += v * v;
        else if (j < 8) g1 += v * v;
        else g2 += v * v;
    }
    cat_h[(size_t)n * 160 + 132] = (_Float16)g0;
    cat_h[(size_t)n * 160 + 133] = (_Float16)g1;
    cat_h[(size_t)n * 160 + 134] = (_Float16)g2;
    for (int f = 135; f < 160; ++f) cat_h[(size_t)n * 160 + f] = (_Float16)0.0f;
}

// ---------------------------------------------------------------------------
// Finish 2: t = cat @ W_int + b_int ; fused output residuals
// ---------------------------------------------------------------------------
__global__ void __launch_bounds__(256) k_finish2(
    const _Float16* __restrict__ cat_h, const _Float16* __restrict__ wint,
    const float* __restrict__ b_int, const float* __restrict__ inv1,
    const float* __restrict__ ev1, float* __restrict__ out_inv,
    float* __restrict__ out_ev) {
    __shared__ float bev[8][16][3];
    const int wave  = threadIdx.x >> 5;
    const int lane  = threadIdx.x & 31;
    const int strip = blockIdx.x * 8 + wave;
    const int n0    = strip * 16;
    if (n0 >= NN) return;

    v16h aa[5];
#pragma unroll
    for (int ks = 0; ks < 5; ++ks)
        aa[ks] = load_a16(cat_h + (size_t)n0 * 160 + ks * 32, 160);

#pragma unroll
    for (int t = 0; t < 9; ++t) {
        v8f acc = {};
#pragma unroll
        for (int ks = 0; ks < 5; ++ks)
            acc = wmma16(aa[ks], load_b_swz(wint + (ks * 9 + t) * 512), acc);
        store_d_guard_col(acc, t * 16, FF, [&](int r, int col, float v) {
            out_inv[(n0 + r) * FF + col] = inv1[(n0 + r) * FF + col] + v + b_int[col];
        });
        // b_ev columns (132..134) -> per-wave LDS (only tile 8 hits this)
        if (t == 8) {
            const int col = t * 16 + (lane & 15);
            const int rb  = (lane >> 4) << 3;
            if (col >= FF && col < FF + 3) {
#pragma unroll
                for (int i = 0; i < 8; ++i)
                    bev[wave][rb + i][col - FF] = acc[i] + b_int[col];
            }
        }
    }
    // same-wave LDS producer/consumer; DS ops from one wave are in-order
    for (int i = lane; i < 16 * EVD; i += 32) {
        int r = i / EVD, j = i - r * EVD;
        int n = n0 + r;
        int g = (j < 3) ? 0 : (j < 8) ? 1 : 2;
        out_ev[n * EVD + j] = ev1[n * EVD + j] * (1.0f + bev[wave][r][g]);
    }
}

// ---------------------------------------------------------------------------
// Host launcher
// ---------------------------------------------------------------------------
extern "C" void kernel_launch(void* const* d_in, const int* in_sizes, int n_in,
                              void* d_out, int out_size, void* d_ws, size_t ws_size,
                              hipStream_t stream) {
    (void)in_sizes; (void)n_in; (void)out_size; (void)ws_size;

    const float* inv_features = (const float*)d_in[0];
    const float* ev_features  = (const float*)d_in[1];
    const float* rbf          = (const float*)d_in[2];
    const float* sh_vectors   = (const float*)d_in[3];
    const float* cutoffs      = (const float*)d_in[4];
    const int*   senders      = (const int*)d_in[5];
    const int*   receivers    = (const int*)d_in[6];
    const float* Wq_inv       = (const float*)d_in[7];
    const float* Wk_inv       = (const float*)d_in[8];
    const float* Wv_inv       = (const float*)d_in[9];
    const float* Wq_ev        = (const float*)d_in[10];
    const float* Wk_ev        = (const float*)d_in[11];
    const float* fi_rW1 = (const float*)d_in[12];
    const float* fi_rb1 = (const float*)d_in[13];
    const float* fi_rW2 = (const float*)d_in[14];
    const float* fi_rb2 = (const float*)d_in[15];
    const float* fi_eW1 = (const float*)d_in[16];
    const float* fi_eb1 = (const float*)d_in[17];
    const float* fi_eW2 = (const float*)d_in[18];
    const float* fi_eb2 = (const float*)d_in[19];
    const float* fe_rW1 = (const float*)d_in[20];
    const float* fe_rb1 = (const float*)d_in[21];
    const float* fe_rW2 = (const float*)d_in[22];
    const float* fe_rb2 = (const float*)d_in[23];
    const float* fe_eW1 = (const float*)d_in[24];
    const float* fe_eb1 = (const float*)d_in[25];
    const float* fe_eW2 = (const float*)d_in[26];
    const float* fe_eb2 = (const float*)d_in[27];
    const float* W_int  = (const float*)d_in[28];
    const float* b_int  = (const float*)d_in[29];

    // ---- workspace carve ----
    _Float16* h16 = (_Float16*)d_ws;
    size_t    off = 0;
    _Float16* w_fi = h16 + off;       off += EW_BLK;
    _Float16* w_fe = h16 + off;       off += EW_BLK;
    _Float16* w_qkv_inv = h16 + off;  off += (size_t)12 * 64 * 48;
    _Float16* w_qk_ev   = h16 + off;  off += (size_t)6 * 64 * 48;
    _Float16* w_int     = h16 + off;  off += (size_t)160 * 144;
    _Float16* cat_h     = h16 + off;  off += (size_t)NN * 160;
    _Float16* xi_inv    = h16 + off;  off += (size_t)NN * 256;
    _Float16* xi_ev     = h16 + off;  off += (size_t)NN * 192;
    float* f32 = (float*)(((uintptr_t)(h16 + off) + 255) & ~(uintptr_t)255);
    size_t fo = 0;
    float* q_inv = f32 + fo; fo += (size_t)NN * FF;
    float* k_inv = f32 + fo; fo += (size_t)NN * FF;
    float* v_inv = f32 + fo; fo += (size_t)NN * FF;
    float* q_ev  = f32 + fo; fo += (size_t)NN * FF;
    float* k_ev  = f32 + fo; fo += (size_t)NN * FF;
    float* d_inv = f32 + fo; fo += (size_t)NN * FF;   // accumulators (contiguous
    float* d_ev  = f32 + fo; fo += (size_t)NN * EVD;  //  so one zero pass covers both)
    float* inv1  = f32 + fo; fo += (size_t)NN * FF;
    float* ev1   = f32 + fo; fo += (size_t)NN * EVD;

    float* out_inv = (float*)d_out;
    float* out_ev  = out_inv + (size_t)NN * FF;

    // ---- 1) pack weights (B-swizzled f16) and padded node features ----
    auto pk = [&](const float* s, _Float16* d, int Ks, int Ns, int KT, int NT,
                  int batch, int bs) {
        int tot = batch * KT * NT * 512;
        k_pack_b_swz<<<(tot + 255) / 256, 256, 0, stream>>>(s, d, Ks, Ns, KT, NT,
                                                            batch, bs);
    };
    pk(fi_rW1, w_fi + EW_RW1_OFF, 32, 132, 1, 9, 1, 0);
    pk(fi_rW2, w_fi + EW_RW2_OFF, 132, 132, 5, 9, 1, 0);
    pk(fi_eW1, w_fi + EW_EW1_OFF, 3, 33, 1, 3, 1, 0);
    pk(fi_eW2, w_fi + EW_EW2_OFF, 33, 132, 2, 9, 1, 0);
    pk(fe_rW1, w_fe + EW_RW1_OFF, 32, 132, 1, 9, 1, 0);
    pk(fe_rW2, w_fe + EW_RW2_OFF, 132, 132, 5, 9, 1, 0);
    pk(fe_eW1, w_fe + EW_EW1_OFF, 3, 33, 1, 3, 1, 0);
    pk(fe_eW2, w_fe + EW_EW2_OFF, 33, 132, 2, 9, 1, 0);
    pk(Wq_inv, w_qkv_inv + 0 * 4 * 3072, 33, 33, 2, 3, 4, 33 * 33);
    pk(Wk_inv, w_qkv_inv + 1 * 4 * 3072, 33, 33, 2, 3, 4, 33 * 33);
    pk(Wv_inv, w_qkv_inv + 2 * 4 * 3072, 33, 33, 2, 3, 4, 33 * 33);
    pk(Wq_ev, w_qk_ev + 0 * 3 * 3072, 44, 44, 2, 3, 3, 44 * 44);
    pk(Wk_ev, w_qk_ev + 1 * 3 * 3072, 44, 44, 2, 3, 3, 44 * 44);
    pk(W_int, w_int, 135, 135, 5, 9, 1, 0);
    k_pack_x<<<(NN * 256 + 255) / 256, 256, 0, stream>>>(inv_features, xi_inv, 4, 33);
    k_pack_x<<<(NN * 192 + 255) / 256, 256, 0, stream>>>(inv_features, xi_ev, 3, 44);

    // ---- 2) zero segment-sum accumulators (d_inv, d_ev contiguous) ----
    k_zero<<<1024, 256, 0, stream>>>(d_inv, (long long)NN * (FF + EVD));

    // ---- 3) node projections ----
    {
        int strips = NN / 16;  // NN % 16 == 0
        int blocks = (strips + 7) / 8;
        k_node_proj<<<blocks, 256, 0, stream>>>(xi_inv, xi_ev, w_qkv_inv, w_qk_ev,
                                                q_inv, k_inv, v_inv, q_ev, k_ev);
    }

    // ---- 4) edge kernels (dynamic LDS ~94KB: legal, 320KB/WGP on CDNA5) ----
    (void)hipFuncSetAttribute(reinterpret_cast<const void*>(k_edge_inv),
                              hipFuncAttributeMaxDynamicSharedMemorySize,
                              (int)sizeof(EdgeLds));
    (void)hipFuncSetAttribute(reinterpret_cast<const void*>(k_edge_ev),
                              hipFuncAttributeMaxDynamicSharedMemorySize,
                              (int)sizeof(EdgeLds));
    k_edge_inv<<<960, 128, sizeof(EdgeLds), stream>>>(
        rbf, cutoffs, senders, receivers, ev_features, w_fi, fi_rb1, fi_rb2, fi_eb1,
        fi_eb2, q_inv, k_inv, v_inv, d_inv);
    k_edge_ev<<<960, 128, sizeof(EdgeLds), stream>>>(
        rbf, cutoffs, senders, receivers, ev_features, sh_vectors, w_fe, fe_rb1,
        fe_rb2, fe_eb1, fe_eb2, q_ev, k_ev, d_ev);

    // ---- 5) residuals + concat staging ----
    k_finish1a<<<(NN * FF + 255) / 256, 256, 0, stream>>>(inv_features, d_inv, inv1,
                                                          cat_h);
    k_finish1b<<<(NN + 255) / 256, 256, 0, stream>>>(ev_features, d_ev, ev1, cat_h);

    // ---- 6) W_int GEMM + fused output epilogue ----
    {
        int strips = NN / 16;
        int blocks = (strips + 7) / 8;
        k_finish2<<<blocks, 256, 0, stream>>>(cat_h, w_int, b_int, inv1, ev1, out_inv,
                                              out_ev);
    }
}
